// Model_Recursive_LSTM_v2_73882027425949
// MI455X (gfx1250) — matrix operand order repaired and checked by reading, not compiled, not run on hardware
//
#include <hip/hip_runtime.h>
#include <hip/hip_bf16.h>

// ---------------------------------------------------------------------------
// CDNA5 (gfx1250) recursive tree-LSTM implementation.
// GEMMs: v_wmma_f32_16x16x32_bf16 (wave32, f32 accumulate), double-buffered
// LDS tiles fed by GLOBAL_LOAD_ASYNC_TO_LDS_B128 (ASYNCcnt-tracked).
// ---------------------------------------------------------------------------

typedef __attribute__((ext_vector_type(16))) __bf16 v16bf;
typedef __attribute__((ext_vector_type(8)))  float  v8f;

__device__ __forceinline__ unsigned short f2bf(float f) {
  unsigned int u = __float_as_uint(f);
  unsigned int r = (u + 0x7FFFu + ((u >> 16) & 1u)) >> 16;   // RNE
  return (unsigned short)r;
}
__device__ __forceinline__ float bf2f(unsigned short h) {
  return __uint_as_float(((unsigned int)h) << 16);
}
__device__ __forceinline__ float sigm_(float x) { return 1.f / (1.f + __expf(-x)); }
__device__ __forceinline__ float tanh_(float x) {
  x = fminf(fmaxf(x, -15.f), 15.f);
  float e = __expf(2.f * x);
  return (e - 1.f) / (e + 1.f);
}
__device__ __forceinline__ float elu_(float x) { return x > 0.f ? x : (__expf(x) - 1.f); }

// ---------------------------------------------------------------------------
// Generic bf16 GEMM: out = act( X[M,K] * W[N,K]^T + bias + addC )
// Block tile 128x128, 256 threads (8 waves). Wave w: rows (w&3)*32, cols (w>>2)*64.
// Per wave: 2x4 accumulators -> 8 WMMAs per 32-wide K step.
// Double-buffered LDS tiles (stride 40 halfwords) filled by async-to-LDS loads.
// Requires: M%128==0, N%128==0, K%32==0 (guaranteed by the model shapes).
// ---------------------------------------------------------------------------
__global__ __launch_bounds__(256) void gemm_wmma(
    const unsigned short* __restrict__ X,     // [M,K] bf16 row-major
    const unsigned short* __restrict__ Wt,    // [N,K] bf16 row-major (torch Linear weight)
    const float* __restrict__ bias,           // [N] fp32 or null
    const float* __restrict__ addC,           // [M,N] fp32 or null
    float* __restrict__ outF,                 // [M,N] fp32 or null
    unsigned short* __restrict__ outH,        // [M,N] bf16 or null
    int M, int N, int K, int act)
{
  __shared__ unsigned short Xs[2][128 * 40];
  __shared__ unsigned short Ws[2][128 * 40];

  const int tid  = threadIdx.x;
  const int lane = tid & 31;
  const int wave = tid >> 5;
  const int lan  = lane & 15;
  const int half = lane >> 4;
  const int m0   = (wave & 3) * 32;    // wave's M offset in block tile
  const int n0   = (wave >> 2) * 64;   // wave's N offset in block tile
  const int blockN = blockIdx.x * 128;
  const int blockM = blockIdx.y * 128;

  // cooperative-load mapping: 128 rows x 4 segs (8 halves each) per matrix;
  // thread covers chunk tid (rows 0..63) and tid+256 (rows 64..127)
  const int crow = tid >> 2;
  const int cseg = tid & 3;

  v8f acc[2][4];
#pragma unroll
  for (int a = 0; a < 2; ++a)
#pragma unroll
    for (int b = 0; b < 4; ++b)
#pragma unroll
      for (int i = 0; i < 8; ++i) acc[a][b][i] = 0.f;

  // Issue one K-tile (128 rows x 32 halves of X and W) as async-to-LDS loads.
  // 4 async instructions per wave per call -> ASYNCcnt batch size 4.
  auto issue = [&](int buf, int kb) {
#pragma unroll
    for (int hh = 0; hh < 2; ++hh) {
      const int row = crow + hh * 64;
      const unsigned long long gx =
          (unsigned long long)(uintptr_t)(X  + (size_t)(blockM + row) * K + kb + cseg * 8);
      const unsigned long long gw =
          (unsigned long long)(uintptr_t)(Wt + (size_t)(blockN + row) * K + kb + cseg * 8);
      const unsigned int lx = (unsigned int)(uintptr_t)&Xs[buf][row * 40 + cseg * 8];
      const unsigned int lw = (unsigned int)(uintptr_t)&Ws[buf][row * 40 + cseg * 8];
      asm volatile("global_load_async_to_lds_b128 %0, %1, off" :: "v"(lx), "v"(gx) : "memory");
      asm volatile("global_load_async_to_lds_b128 %0, %1, off" :: "v"(lw), "v"(gw) : "memory");
    }
  };

  const int nk = K >> 5;
  issue(0, 0);
  for (int i = 0; i < nk; ++i) {
    const int buf = i & 1;
    if (i + 1 < nk) {
      issue(buf ^ 1, (i + 1) << 5);                       // prefetch next tile
      asm volatile("s_wait_asynccnt 4" ::: "memory");     // current tile landed; next in flight
    } else {
      asm volatile("s_wait_asynccnt 0" ::: "memory");
    }
    __syncthreads();

    const unsigned int* Xs32 = (const unsigned int*)Xs[buf];
    const unsigned int* Ws32 = (const unsigned int*)Ws[buf];
    union Frag { v16bf v; unsigned int u[8]; };
    Frag a[2], b[4];

    // A (16x32 bf16): lane holds row; word j -> K pair (j>>2)*16 + half*8 + (j&3)*2
#pragma unroll
    for (int mi = 0; mi < 2; ++mi) {
      const int mrow = m0 + mi * 16 + lan;
#pragma unroll
      for (int j = 0; j < 8; ++j)
        a[mi].u[j] = Xs32[mrow * 20 + (j >> 2) * 8 + half * 4 + (j & 3)];
    }
    // B (32x16 bf16): lane holds column; word j -> K pair (j>>2)*16 + half*8 + (j&3)*2
#pragma unroll
    for (int ni = 0; ni < 4; ++ni) {
      const int nrow = n0 + ni * 16 + lan;
#pragma unroll
      for (int j = 0; j < 8; ++j)
        b[ni].u[j] = Ws32[nrow * 20 + (j >> 2) * 8 + half * 4 + (j & 3)];
    }

#pragma unroll
    for (int mi = 0; mi < 2; ++mi)
#pragma unroll
      for (int ni = 0; ni < 4; ++ni)
        acc[mi][ni] = __builtin_amdgcn_wmma_f32_16x16x32_bf16(
            false, a[mi].v, false, b[ni].v, (short)0, acc[mi][ni], false, false);

    __syncthreads();   // protect buf from batch i+2 async writes
  }

  // epilogue: C/D layout -> lane col = lan, vgpr r row = r + 8*half
#pragma unroll
  for (int mi = 0; mi < 2; ++mi) {
#pragma unroll
    for (int ni = 0; ni < 4; ++ni) {
      const int gn = blockN + n0 + ni * 16 + lan;
      const float bv = bias ? bias[gn] : 0.f;
#pragma unroll
      for (int r = 0; r < 8; ++r) {
        const int gm = blockM + m0 + mi * 16 + r + half * 8;
        const size_t o = (size_t)gm * N + gn;
        float v = acc[mi][ni][r] + bv;
        if (addC) v += addC[o];
        if (act)  v = elu_(v);
        if (outF) outF[o] = v;
        if (outH) outH[o] = f2bf(v);
      }
    }
  }
}

// ---------------------------------------------------------------------------
// Pointwise / plumbing kernels
// ---------------------------------------------------------------------------
__global__ void pack_bf16_k(const float* __restrict__ s, unsigned short* __restrict__ d, int n) {
  int i = blockIdx.x * blockDim.x + threadIdx.x;
  if (i < n) d[i] = f2bf(s[i]);
}
__global__ void addvec_k(const float* __restrict__ a, const float* __restrict__ b,
                         float* __restrict__ o, int n) {
  int i = blockIdx.x * blockDim.x + threadIdx.x;
  if (i < n) o[i] = a[i] + b[i];
}
__global__ void zero_u16_k(unsigned short* p, int n) {
  int i = blockIdx.x * blockDim.x + threadIdx.x;
  if (i < n) p[i] = 0;
}
__global__ void zero_f32_k(float* p, int n) {
  int i = blockIdx.x * blockDim.x + threadIdx.x;
  if (i < n) p[i] = 0.f;
}

// LSTM pointwise cell: gates [rows,512] (i,f,g,o), c [rows,128] fp32, h [rows,128] bf16
__global__ void lstm_cell_k(const float* __restrict__ gates, float* __restrict__ c,
                            unsigned short* __restrict__ h, int rows) {
  int i = blockIdx.x * blockDim.x + threadIdx.x;
  if (i >= rows * 128) return;
  int m = i >> 7, j = i & 127;
  const float* g = gates + (size_t)m * 512;
  float si = sigm_(g[j]);
  float sf = sigm_(g[128 + j]);
  float tg = tanh_(g[256 + j]);
  float so = sigm_(g[384 + j]);
  float cv = sf * c[i] + si * tg;
  c[i] = cv;
  h[i] = f2bf(so * tanh_(cv));
}

// Gather comps_emb rows for a leaf into t-major [16][2048][128]
__global__ void leaf_gather_k(const unsigned short* __restrict__ emb,
                              unsigned short* __restrict__ out, int leaf) {
  int i = blockIdx.x * blockDim.x + threadIdx.x;
  if (i >= 16 * 2048 * 128) return;
  int c = i & 127;
  int r = i >> 7;            // t*2048 + b
  int b = r & 2047;
  int t = r >> 11;
  out[i] = emb[((size_t)((b << 6) + (leaf << 4) + t)) * 128 + c];
}

// Gather leaf states [4][2048][128] into [t][mid][2048][128] (t-major for the node LSTM)
__global__ void mid_gather_k(const unsigned short* __restrict__ leafS,
                             unsigned short* __restrict__ out) {
  int i = blockIdx.x * blockDim.x + threadIdx.x;
  if (i >= 8192 * 128) return;
  int c = i & 127;
  int r = i >> 7;            // t*4096 + m*2048 + b
  int b = r & 2047;
  int mt = r >> 11;          // 2t + m
  int t = mt >> 1, m = mt & 1;
  out[i] = leafS[((size_t)((m * 2 + t) * 2048 + b)) * 128 + c];
}

// Build concat input [rows,384] = [nodes_h | comps_h | lemb(li0+node)], rows = nodes*2048
__global__ void concat_build_k(const unsigned short* __restrict__ nodesH,  // null => broadcast nb
                               const unsigned short* __restrict__ nb,      // [128]
                               const unsigned short* __restrict__ compsH,  // null => broadcast cb
                               const unsigned short* __restrict__ cb,      // [128]
                               const unsigned short* __restrict__ lemb,    // [B*7,128]
                               unsigned short* __restrict__ out, int rows, int li0) {
  int i = blockIdx.x * blockDim.x + threadIdx.x;
  if (i >= rows * 384) return;
  int r = i / 384;
  int col = i - r * 384;
  int b = r & 2047;
  int node = r >> 11;
  unsigned short v;
  if (col < 128) {
    v = nodesH ? nodesH[(size_t)r * 128 + col] : nb[col];
  } else if (col < 256) {
    int c2 = col - 128;
    v = compsH ? compsH[(size_t)r * 128 + c2] : cb[c2];
  } else {
    int c2 = col - 256;
    v = lemb[((size_t)(b * 7 + li0 + node)) * 128 + c2];
  }
  out[i] = v;
}

// out[r] = elu( dot(x[r,:128], w) + b ); one wave per row
__global__ void predict_k(const unsigned short* __restrict__ x, const unsigned short* __restrict__ wv,
                          const float* __restrict__ bscalar, float* __restrict__ out, int rows) {
  int r = blockIdx.x * (blockDim.x >> 5) + (threadIdx.x >> 5);
  int lane = threadIdx.x & 31;
  if (r >= rows) return;
  float s = 0.f;
  for (int j = lane; j < 128; j += 32)
    s += bf2f(x[(size_t)r * 128 + j]) * bf2f(wv[j]);
  for (int o = 16; o > 0; o >>= 1) s += __shfl_down(s, o, 32);
  if (lane == 0) out[r] = elu_(s + bscalar[0]);
}

// ---------------------------------------------------------------------------
// Host orchestration
// ---------------------------------------------------------------------------
static inline int cdiv(int a, int b) { return (a + b - 1) / b; }

extern "C" void kernel_launch(void* const* d_in, const int* in_sizes, int n_in,
                              void* d_out, int out_size, void* d_ws, size_t ws_size,
                              hipStream_t stream) {
  (void)n_in; (void)out_size; (void)ws_size; (void)in_sizes;

  const float* comps_tensor = (const float*)d_in[0];   // [2048,64,256]
  const float* loops_tensor = (const float*)d_in[1];   // [2048,7,32]

  static const int compN[5] = {128, 256, 256, 256, 128};
  static const int compK[5] = {256, 128, 256, 256, 256};
  static const int loopN[7] = {256, 128, 256, 256, 256, 128, 128};
  static const int loopK[7] = {32, 256, 128, 256, 256, 256, 128};
  static const int catN[4]  = {128, 256, 256, 128};
  static const int catK[4]  = {384, 128, 256, 256};
  static const int regN[3]  = {128, 256, 128};
  static const int regK[3]  = {128, 128, 256};

  const float *compW[5], *compB[5], *loopW[7], *loopB[7], *catW[4], *catB[4], *regW[3], *regB[3];
  int ii = 2;
  for (int i = 0; i < 5; ++i) { compW[i] = (const float*)d_in[ii++]; compB[i] = (const float*)d_in[ii++]; }
  for (int i = 0; i < 7; ++i) { loopW[i] = (const float*)d_in[ii++]; loopB[i] = (const float*)d_in[ii++]; }
  for (int i = 0; i < 4; ++i) { catW[i]  = (const float*)d_in[ii++]; catB[i]  = (const float*)d_in[ii++]; }
  for (int i = 0; i < 3; ++i) { regW[i]  = (const float*)d_in[ii++]; regB[i]  = (const float*)d_in[ii++]; }
  const float* predW = (const float*)d_in[ii++];
  const float* predB = (const float*)d_in[ii++];
  const float* WihC = (const float*)d_in[ii++];
  const float* WhhC = (const float*)d_in[ii++];
  const float* bihC = (const float*)d_in[ii++];
  const float* bhhC = (const float*)d_in[ii++];
  const float* WihN = (const float*)d_in[ii++];
  const float* WhhN = (const float*)d_in[ii++];
  const float* bihN = (const float*)d_in[ii++];
  const float* bhhN = (const float*)d_in[ii++];
  const float* noComps = (const float*)d_in[ii++];   // [1,1,128]
  const float* noNodes = (const float*)d_in[ii++];   // [1,1,128]

  // ---- workspace bump allocator ----
  char* base = (char*)d_ws;
  size_t off = 0;
  auto alloc = [&](size_t bytes) -> void* {
    void* p = base + off;
    off += (bytes + 255) & ~(size_t)255;
    return p;
  };

  unsigned short* wCompBf[5]; for (int i = 0; i < 5; ++i) wCompBf[i] = (unsigned short*)alloc((size_t)compN[i] * compK[i] * 2);
  unsigned short* wLoopBf[7]; for (int i = 0; i < 7; ++i) wLoopBf[i] = (unsigned short*)alloc((size_t)loopN[i] * loopK[i] * 2);
  unsigned short* wCatBf[4];  for (int i = 0; i < 4; ++i) wCatBf[i]  = (unsigned short*)alloc((size_t)catN[i] * catK[i] * 2);
  unsigned short* wRegBf[3];  for (int i = 0; i < 3; ++i) wRegBf[i]  = (unsigned short*)alloc((size_t)regN[i] * regK[i] * 2);
  unsigned short* wIhC = (unsigned short*)alloc(512 * 128 * 2);
  unsigned short* wHhC = (unsigned short*)alloc(512 * 128 * 2);
  unsigned short* wIhN = (unsigned short*)alloc(512 * 128 * 2);
  unsigned short* wHhN = (unsigned short*)alloc(512 * 128 * 2);
  unsigned short* wPredBf = (unsigned short*)alloc(128 * 2);
  unsigned short* nbC = (unsigned short*)alloc(128 * 2);
  unsigned short* nbN = (unsigned short*)alloc(128 * 2);
  float* cbC = (float*)alloc(512 * 4);
  float* cbN = (float*)alloc(512 * 4);

  unsigned short* actA = (unsigned short*)alloc((size_t)131072 * 256 * 2);   // 67 MB
  unsigned short* actB = (unsigned short*)alloc((size_t)131072 * 256 * 2);   // 67 MB (doubles as fp32 xg)
  unsigned short* compsEmb = (unsigned short*)alloc((size_t)131072 * 128 * 2);
  unsigned short* loopsX = (unsigned short*)alloc((size_t)14336 * 32 * 2);
  unsigned short* lembA  = (unsigned short*)alloc((size_t)14336 * 128 * 2);
  unsigned short* leafH  = (unsigned short*)alloc((size_t)4 * 2048 * 128 * 2);
  unsigned short* leafS  = (unsigned short*)alloc((size_t)4 * 2048 * 128 * 2);
  unsigned short* midH   = (unsigned short*)alloc((size_t)4096 * 128 * 2);
  unsigned short* midS   = (unsigned short*)alloc((size_t)2 * 2048 * 128 * 2);
  unsigned short* rootH  = (unsigned short*)alloc((size_t)2048 * 128 * 2);
  unsigned short* rootS  = (unsigned short*)alloc((size_t)2048 * 128 * 2);
  float* gates = (float*)alloc((size_t)4096 * 512 * 4);
  float* cbuf  = (float*)alloc((size_t)4096 * 128 * 4);

  auto pack = [&](const float* s, unsigned short* d, int n) {
    pack_bf16_k<<<cdiv(n, 256), 256, 0, stream>>>(s, d, n);
  };
  auto gemm = [&](const unsigned short* X, const unsigned short* W, const float* bias,
                  const float* addC, float* outF, unsigned short* outH,
                  int M, int N, int K, int act) {
    dim3 g(N / 128, M / 128);
    gemm_wmma<<<g, 256, 0, stream>>>(X, W, bias, addC, outF, outH, M, N, K, act);
  };

  // ---- 1. pack weights & constants to bf16 ----
  for (int i = 0; i < 5; ++i) pack(compW[i], wCompBf[i], compN[i] * compK[i]);
  for (int i = 0; i < 7; ++i) pack(loopW[i], wLoopBf[i], loopN[i] * loopK[i]);
  for (int i = 0; i < 4; ++i) pack(catW[i],  wCatBf[i],  catN[i] * catK[i]);
  for (int i = 0; i < 3; ++i) pack(regW[i],  wRegBf[i],  regN[i] * regK[i]);
  pack(WihC, wIhC, 512 * 128); pack(WhhC, wHhC, 512 * 128);
  pack(WihN, wIhN, 512 * 128); pack(WhhN, wHhN, 512 * 128);
  pack(predW, wPredBf, 128);
  pack(noComps, nbC, 128);
  pack(noNodes, nbN, 128);
  addvec_k<<<2, 256, 0, stream>>>(bihC, bhhC, cbC, 512);
  addvec_k<<<2, 256, 0, stream>>>(bihN, bhhN, cbN, 512);

  // ---- 2. comps embedding MLP (M=131072) ----
  pack(comps_tensor, actA, 131072 * 256);
  gemm(actA, wCompBf[0], compB[0], nullptr, nullptr, actB,     131072, 128, 256, 1);
  gemm(actB, wCompBf[1], compB[1], nullptr, nullptr, actA,     131072, 256, 128, 1);
  gemm(actA, wCompBf[2], compB[2], nullptr, nullptr, actB,     131072, 256, 256, 1);
  gemm(actB, wCompBf[3], compB[3], nullptr, nullptr, actA,     131072, 256, 256, 1);
  gemm(actA, wCompBf[4], compB[4], nullptr, nullptr, compsEmb, 131072, 128, 256, 1);

  // ---- 3. loop MLP for all 7 loop nodes (M=14336) ----
  pack(loops_tensor, loopsX, 14336 * 32);
  gemm(loopsX, wLoopBf[0], loopB[0], nullptr, nullptr, actA,  14336, 256, 32, 1);
  gemm(actA,   wLoopBf[1], loopB[1], nullptr, nullptr, actB,  14336, 128, 256, 1);
  gemm(actB,   wLoopBf[2], loopB[2], nullptr, nullptr, actA,  14336, 256, 128, 1);
  gemm(actA,   wLoopBf[3], loopB[3], nullptr, nullptr, actB,  14336, 256, 256, 1);
  gemm(actB,   wLoopBf[4], loopB[4], nullptr, nullptr, actA,  14336, 256, 256, 1);
  gemm(actA,   wLoopBf[5], loopB[5], nullptr, nullptr, actB,  14336, 128, 256, 1);
  gemm(actB,   wLoopBf[6], loopB[6], nullptr, nullptr, lembA, 14336, 128, 128, 1);

  // ---- 4. leaf LSTMs over comps (4 leaves, T=16) ----
  float* xg = (float*)actB;
  for (int l = 0; l < 4; ++l) {
    leaf_gather_k<<<cdiv(16 * 2048 * 128, 256), 256, 0, stream>>>(compsEmb, actA, l);
    gemm(actA, wIhC, cbC, nullptr, xg, nullptr, 32768, 512, 128, 0);   // xg[16][2048][512]
    unsigned short* h = leafH + (size_t)l * 2048 * 128;
    zero_u16_k<<<cdiv(2048 * 128, 256), 256, 0, stream>>>(h, 2048 * 128);
    zero_f32_k<<<cdiv(2048 * 128, 256), 256, 0, stream>>>(cbuf, 2048 * 128);
    for (int t = 0; t < 16; ++t) {
      gemm(h, wHhC, nullptr, xg + (size_t)t * 2048 * 512, gates, nullptr, 2048, 512, 128, 0);
      lstm_cell_k<<<cdiv(2048 * 128, 256), 256, 0, stream>>>(gates, cbuf, h, 2048);
    }
  }

  // ---- 5. leaf concat MLPs (batched over 4 leaves, M=8192, li = 3..6) ----
  concat_build_k<<<cdiv(8192 * 384, 256), 256, 0, stream>>>(nullptr, nbN, leafH, nbC, lembA, actA, 8192, 3);
  gemm(actA, wCatBf[0], catB[0], nullptr, nullptr, actB,  8192, 128, 384, 1);
  gemm(actB, wCatBf[1], catB[1], nullptr, nullptr, actA,  8192, 256, 128, 1);
  gemm(actA, wCatBf[2], catB[2], nullptr, nullptr, actB,  8192, 256, 256, 1);
  gemm(actB, wCatBf[3], catB[3], nullptr, nullptr, leafS, 8192, 128, 256, 1);

  // ---- 6. mid-node LSTMs over children (2 mids batched, T=2) ----
  mid_gather_k<<<cdiv(8192 * 128, 256), 256, 0, stream>>>(leafS, actA);
  gemm(actA, wIhN, cbN, nullptr, xg, nullptr, 8192, 512, 128, 0);      // xg[2][4096][512]
  zero_u16_k<<<cdiv(4096 * 128, 256), 256, 0, stream>>>(midH, 4096 * 128);
  zero_f32_k<<<cdiv(4096 * 128, 256), 256, 0, stream>>>(cbuf, 4096 * 128);
  for (int t = 0; t < 2; ++t) {
    gemm(midH, wHhN, nullptr, xg + (size_t)t * 4096 * 512, gates, nullptr, 4096, 512, 128, 0);
    lstm_cell_k<<<cdiv(4096 * 128, 256), 256, 0, stream>>>(gates, cbuf, midH, 4096);
  }
  // mid concat MLPs (M=4096, li = 1..2)
  concat_build_k<<<cdiv(4096 * 384, 256), 256, 0, stream>>>(midH, nbN, nullptr, nbC, lembA, actA, 4096, 1);
  gemm(actA, wCatBf[0], catB[0], nullptr, nullptr, actB, 4096, 128, 384, 1);
  gemm(actB, wCatBf[1], catB[1], nullptr, nullptr, actA, 4096, 256, 128, 1);
  gemm(actA, wCatBf[2], catB[2], nullptr, nullptr, actB, 4096, 256, 256, 1);
  gemm(actB, wCatBf[3], catB[3], nullptr, nullptr, midS, 4096, 128, 256, 1);

  // ---- 7. root LSTM over the 2 mid states (T=2; midS is already t-major) ----
  gemm(midS, wIhN, cbN, nullptr, xg, nullptr, 4096, 512, 128, 0);      // xg[2][2048][512]
  zero_u16_k<<<cdiv(2048 * 128, 256), 256, 0, stream>>>(rootH, 2048 * 128);
  zero_f32_k<<<cdiv(2048 * 128, 256), 256, 0, stream>>>(cbuf, 2048 * 128);
  for (int t = 0; t < 2; ++t) {
    gemm(rootH, wHhN, nullptr, xg + (size_t)t * 2048 * 512, gates, nullptr, 2048, 512, 128, 0);
    lstm_cell_k<<<cdiv(2048 * 128, 256), 256, 0, stream>>>(gates, cbuf, rootH, 2048);
  }
  // root concat MLP (M=2048, li = 0)
  concat_build_k<<<cdiv(2048 * 384, 256), 256, 0, stream>>>(rootH, nbN, nullptr, nbC, lembA, actA, 2048, 0);
  gemm(actA, wCatBf[0], catB[0], nullptr, nullptr, actB,  2048, 128, 384, 1);
  gemm(actB, wCatBf[1], catB[1], nullptr, nullptr, actA,  2048, 256, 128, 1);
  gemm(actA, wCatBf[2], catB[2], nullptr, nullptr, actB,  2048, 256, 256, 1);
  gemm(actB, wCatBf[3], catB[3], nullptr, nullptr, rootS, 2048, 128, 256, 1);

  // ---- 8. regression MLP + predict ----
  gemm(rootS, wRegBf[0], regB[0], nullptr, nullptr, actA, 2048, 128, 128, 1);
  gemm(actA,  wRegBf[1], regB[1], nullptr, nullptr, actB, 2048, 256, 128, 1);
  gemm(actB,  wRegBf[2], regB[2], nullptr, nullptr, actA, 2048, 128, 256, 1);
  predict_k<<<cdiv(2048, 8), 256, 0, stream>>>(actA, wPredBf, predB, (float*)d_out, 2048);
}
